// SpatialAttentionModule_3384434229772
// MI455X (gfx1250) — matrix operand order repaired and checked by reading, not compiled, not run on hardware
//
#include <hip/hip_runtime.h>
#include <hip/hip_bf16.h>

typedef __attribute__((ext_vector_type(16))) __bf16 v16bf;
typedef __attribute__((ext_vector_type(8)))  float  v8f;

union Frag {
  v16bf v;
  uint4 q[2];     // two 128-bit halves (v=0..3, v=4..7)
  unsigned u[8];
};

#define WMMA_BF16(A, B, C) \
  __builtin_amdgcn_wmma_f32_16x16x32_bf16(false, (A), false, (B), (short)0, (C), false, false)

__device__ __forceinline__ unsigned short f2bf(float f) {
  unsigned u = __builtin_bit_cast(unsigned, f);
  unsigned r = u + 0x7FFFu + ((u >> 16) & 1u);   // round-to-nearest-even
  return (unsigned short)(r >> 16);
}

// ---------------------------------------------------------------------------
// Prep: weights OIHW fp32 -> bf16 [tap=9][co=256][ci=256]
// ---------------------------------------------------------------------------
__global__ void pack_w_kernel(const float* __restrict__ w, unsigned short* __restrict__ Wp) {
  int i = blockIdx.x * 256 + threadIdx.x;          // 9*256*256 = 589824
  if (i >= 9 * 256 * 256) return;
  int ci  = i & 255;
  int co  = (i >> 8) & 255;
  int tap = i >> 16;
  int r = tap / 3, s = tap % 3;
  Wp[i] = f2bf(w[(((size_t)co * 256 + ci) * 3 + r) * 3 + s]);
}

// ---------------------------------------------------------------------------
// Prep: X NCHW fp32 -> padded NHWC bf16 [8][66][66][256], zero halo
// ---------------------------------------------------------------------------
__global__ void pad_x_kernel(const float* __restrict__ X, unsigned short* __restrict__ Xp) {
  size_t i = (size_t)blockIdx.x * 256 + threadIdx.x;   // 8*66*66*256
  if (i >= (size_t)8 * 66 * 66 * 256) return;
  int ci = (int)(i & 255);
  size_t t = i >> 8;
  int xx = (int)(t % 66); t /= 66;
  int yy = (int)(t % 66);
  int b  = (int)(t / 66);
  float v = 0.f;
  if (xx >= 1 && xx <= 64 && yy >= 1 && yy <= 64)
    v = X[(((size_t)b * 256 + ci) * 64 + (yy - 1)) * 64 + (xx - 1)];
  Xp[i] = f2bf(v);
}

// ---------------------------------------------------------------------------
// Conv3x3 as implicit GEMM. One wave -> 16(co) x 64(n) tile (one image row),
// so each weight A-fragment is reused across 4 B-fragments.
// K loop: 9 taps x 8 chunks of 32 ci -> 288 WMMAs / wave.
// layoutNC=1: write bf16 [b][n][256]  (A,B for attention)
// layoutNC=0: write bf16 [b][256][n]  (D for attention)
// ---------------------------------------------------------------------------
__global__ void __launch_bounds__(128) conv_wmma_kernel(
    const unsigned short* __restrict__ Xp,   // [8][66][66][256] bf16
    const unsigned short* __restrict__ Wp,   // [9][256][256] bf16
    const float* __restrict__ bias,          // [256]
    unsigned short* __restrict__ out,        // bf16
    int layoutNC) {
  const int lane = threadIdx.x & 31;
  const int wave = threadIdx.x >> 5;
  const int tile = blockIdx.x * 4 + wave;    // 2048 blocks * 4 waves = 8192 tiles
  const int y  = tile & 63;                  // image row
  const int ct = (tile >> 6) & 15;           // co-tile
  const int b  = tile >> 10;                 // batch
  const int co0 = ct * 16;
  const int half = lane >> 4;
  const int l15  = lane & 15;

  v8f acc[4];
#pragma unroll
  for (int j = 0; j < 4; ++j) acc[j] = (v8f){0.f,0.f,0.f,0.f,0.f,0.f,0.f,0.f};

  for (int tap = 0; tap < 9; ++tap) {
    const int r = tap / 3, s = tap % 3;
    // A-fragment rows: co = co0 + (lane&15)
    const unsigned short* wbase = Wp + (size_t)tap * 65536 + (size_t)(co0 + l15) * 256;
    // B-fragment cols: x = 16*j + (lane&15); padded indices (y+r, x+s)
    const unsigned short* xbase =
        Xp + (((size_t)b * 66 + y + r) * 66 + (l15 + s)) * 256;
#pragma unroll
    for (int kc = 0; kc < 8; ++kc) {
      Frag A;
      // A layout: q0 covers K = kc*32 + 8*half + {0..7}, q1 = +16
      A.q[0] = *(const uint4*)(wbase + kc * 32 + 8 * half);
      A.q[1] = *(const uint4*)(wbase + kc * 32 + 8 * half + 16);
#pragma unroll
      for (int j = 0; j < 4; ++j) {
        Frag B;
        // B layout: q0 covers K = kc*32 + 8*half + {0..7}, q1 = +16
        const unsigned short* xb = xbase + (size_t)j * 16 * 256 + kc * 32 + 8 * half;
        B.q[0] = *(const uint4*)(xb);
        B.q[1] = *(const uint4*)(xb + 16);
        acc[j] = WMMA_BF16(A.v, B.v, acc[j]);
      }
    }
  }

  // Epilogue: +bias, cvt bf16, store. C/D layout: row co = rr+8*half, col n = lane&15
#pragma unroll
  for (int j = 0; j < 4; ++j) {
    const int n = y * 64 + j * 16 + l15;
#pragma unroll
    for (int rr = 0; rr < 8; ++rr) {
      const int co = co0 + rr + 8 * half;
      const unsigned short bv = f2bf(acc[j][rr] + bias[co]);
      const size_t idx = layoutNC ? (((size_t)b * 4096 + n) * 256 + co)
                                  : (((size_t)b * 256 + co) * 4096 + n);
      out[idx] = bv;
    }
  }
}

// ---------------------------------------------------------------------------
// Flash attention: one 32-thread wave per (batch, 16-row query tile).
// out[n,c] = alpha * softmax_m( A[n,:]·B[m,:] ) · D[c,m]
// ---------------------------------------------------------------------------
__global__ void __launch_bounds__(32) attn_flash_kernel(
    const unsigned short* __restrict__ Abuf,  // [8][4096][256] bf16
    const unsigned short* __restrict__ Bbuf,  // [8][4096][256] bf16
    const unsigned short* __restrict__ Dbuf,  // [8][256][4096] bf16
    const float* __restrict__ alphap,
    float* __restrict__ out) {                // [8][256][4096] f32
  __shared__ __align__(16) unsigned short Plds[16 * 32];  // P transpose staging (1 KB)

  const int lane = threadIdx.x & 31;
  const int half = lane >> 4;
  const int l15  = lane & 15;
  const int unit = blockIdx.x;                // 0..2047
  const int b    = unit >> 8;
  const int n0   = (unit & 255) * 16;
  const float alpha = alphap[0];
  const float LOG2E = 1.4426950408889634f;

  // Q: 8 A-fragments (16n x 32c each), resident in VGPRs
  Frag Q[8];
  {
    const unsigned short* qbase = Abuf + ((size_t)b * 4096 + n0 + l15) * 256;
#pragma unroll
    for (int kc = 0; kc < 8; ++kc) {
      Q[kc].q[0] = *(const uint4*)(qbase + kc * 32 + 8 * half);
      Q[kc].q[1] = *(const uint4*)(qbase + kc * 32 + 8 * half + 16);
    }
  }

  v8f O[16];                                  // 16n x 256c accumulator
#pragma unroll
  for (int i = 0; i < 16; ++i) O[i] = (v8f){0.f,0.f,0.f,0.f,0.f,0.f,0.f,0.f};
  float mrow[8], lrow[8];
#pragma unroll
  for (int r = 0; r < 8; ++r) { mrow[r] = -3.0e38f; lrow[r] = 0.f; }

  for (int m0 = 0; m0 < 4096; m0 += 32) {
    // ---- logits: two 16x16 tiles, K = 256 ----
    v8f S0 = (v8f){0.f,0.f,0.f,0.f,0.f,0.f,0.f,0.f};
    v8f S1 = (v8f){0.f,0.f,0.f,0.f,0.f,0.f,0.f,0.f};
    {
      const unsigned short* kb0 = Bbuf + ((size_t)b * 4096 + m0 + l15) * 256;
      const unsigned short* kb1 = kb0 + 16 * 256;   // next 16 m rows
#pragma unroll
      for (int kc = 0; kc < 8; ++kc) {
        Frag B0, B1;
        const int k0 = kc * 32 + 8 * half;
        B0.q[0] = *(const uint4*)(kb0 + k0);
        B0.q[1] = *(const uint4*)(kb0 + k0 + 16);
        B1.q[0] = *(const uint4*)(kb1 + k0);
        B1.q[1] = *(const uint4*)(kb1 + k0 + 16);
        S0 = WMMA_BF16(Q[kc].v, B0.v, S0);
        S1 = WMMA_BF16(Q[kc].v, B1.v, S1);
      }
    }

    // ---- online softmax (row n = r + 8*half lives in one 16-lane group) ----
    float scale[8];
#pragma unroll
    for (int r = 0; r < 8; ++r) {
      float cm = fmaxf(S0[r], S1[r]);
      cm = fmaxf(cm, __shfl_xor(cm, 1));
      cm = fmaxf(cm, __shfl_xor(cm, 2));
      cm = fmaxf(cm, __shfl_xor(cm, 4));
      cm = fmaxf(cm, __shfl_xor(cm, 8));
      const float mn = fmaxf(mrow[r], cm);
      scale[r] = exp2f((mrow[r] - mn) * LOG2E);
      mrow[r]  = mn;
      const float p0 = exp2f((S0[r] - mn) * LOG2E);
      const float p1 = exp2f((S1[r] - mn) * LOG2E);
      S0[r] = p0; S1[r] = p1;
      lrow[r] = lrow[r] * scale[r] + p0 + p1;
    }
#pragma unroll
    for (int ctile = 0; ctile < 16; ++ctile)
#pragma unroll
      for (int r = 0; r < 8; ++r) O[ctile][r] *= scale[r];

    // ---- transpose P (C/D layout -> A layout) via LDS ----
    __syncthreads();                          // fence: previous iteration's reads done
#pragma unroll
    for (int r = 0; r < 8; ++r) {
      const int n = r + 8 * half;
      Plds[n * 32 + l15]      = f2bf(S0[r]);
      Plds[n * 32 + 16 + l15] = f2bf(S1[r]);
    }
    __syncthreads();
    Frag Pa;
    Pa.q[0] = *(const uint4*)(Plds + l15 * 32 + 8 * half);
    Pa.q[1] = *(const uint4*)(Plds + l15 * 32 + 8 * half + 16);

    // ---- O[n,c] += P(16x32m) x D(32m x 16c) for 16 c-tiles ----
    const unsigned short* dbase = Dbuf + ((size_t)b * 256 + l15) * 4096 + m0 + 8 * half;
#pragma unroll
    for (int ctile = 0; ctile < 16; ++ctile) {
      Frag Dv;
      Dv.q[0] = *(const uint4*)(dbase + (size_t)ctile * 16 * 4096);
      Dv.q[1] = *(const uint4*)(dbase + (size_t)ctile * 16 * 4096 + 16);
      O[ctile] = WMMA_BF16(Pa.v, Dv.v, O[ctile]);
    }
  }

  // ---- finalize: row sum reduce, scale by alpha/l, store [b][c][n] fp32 ----
#pragma unroll
  for (int r = 0; r < 8; ++r) {
    float l = lrow[r];
    l += __shfl_xor(l, 1);
    l += __shfl_xor(l, 2);
    l += __shfl_xor(l, 4);
    l += __shfl_xor(l, 8);
    lrow[r] = alpha / l;
  }
#pragma unroll
  for (int ctile = 0; ctile < 16; ++ctile) {
    const int c = ctile * 16 + l15;
    float* obase = out + ((size_t)b * 256 + c) * 4096 + n0;
#pragma unroll
    for (int r = 0; r < 8; ++r)
      obase[r + 8 * half] = O[ctile][r] * lrow[r];
  }
}

// ---------------------------------------------------------------------------
extern "C" void kernel_launch(void* const* d_in, const int* in_sizes, int n_in,
                              void* d_out, int out_size, void* d_ws, size_t ws_size,
                              hipStream_t stream) {
  const float* X  = (const float*)d_in[0];
  const float* w1 = (const float*)d_in[1];
  const float* b1 = (const float*)d_in[2];
  const float* w2 = (const float*)d_in[3];
  const float* b2 = (const float*)d_in[4];
  const float* w3 = (const float*)d_in[5];
  const float* b3 = (const float*)d_in[6];
  const float* alpha = (const float*)d_in[7];
  float* out = (float*)d_out;

  // Workspace layout (bytes)
  const size_t XP_BYTES = (size_t)8 * 66 * 66 * 256 * 2;   // 17,842,176
  const size_t WP_BYTES = (size_t)9 * 256 * 256 * 2;       //  1,179,648 per conv
  const size_t AB_BYTES = (size_t)8 * 4096 * 256 * 2;      // 16,777,216
  size_t off = 0;
  unsigned short* Xp  = (unsigned short*)((char*)d_ws + off); off += XP_BYTES;
  unsigned short* Wp1 = (unsigned short*)((char*)d_ws + off); off += WP_BYTES;
  unsigned short* Wp2 = (unsigned short*)((char*)d_ws + off); off += WP_BYTES;
  unsigned short* Wp3 = (unsigned short*)((char*)d_ws + off); off += WP_BYTES;
  unsigned short* Abuf = (unsigned short*)((char*)d_ws + off); off += AB_BYTES;
  unsigned short* Bbuf = (unsigned short*)((char*)d_ws + off); off += AB_BYTES;
  unsigned short* Dbuf = (unsigned short*)((char*)d_ws + off); off += AB_BYTES;
  if (ws_size < off) return;   // insufficient scratch

  // 1) pack weights to bf16 [tap][co][ci]
  {
    const int n = 9 * 256 * 256;
    const int blocks = (n + 255) / 256;
    pack_w_kernel<<<blocks, 256, 0, stream>>>(w1, Wp1);
    pack_w_kernel<<<blocks, 256, 0, stream>>>(w2, Wp2);
    pack_w_kernel<<<blocks, 256, 0, stream>>>(w3, Wp3);
  }
  // 2) pad+transpose X to NHWC bf16
  {
    const size_t n = (size_t)8 * 66 * 66 * 256;
    pad_x_kernel<<<(unsigned)((n + 255) / 256), 256, 0, stream>>>(X, Xp);
  }
  // 3) three convs (A,B in [b][n][c]; D in [b][c][n])
  conv_wmma_kernel<<<2048, 128, 0, stream>>>(Xp, Wp1, b1, Abuf, 1);
  conv_wmma_kernel<<<2048, 128, 0, stream>>>(Xp, Wp2, b2, Bbuf, 1);
  conv_wmma_kernel<<<2048, 128, 0, stream>>>(Xp, Wp3, b3, Dbuf, 0);
  // 4) fused flash attention + alpha scale
  attn_flash_kernel<<<2048, 32, 0, stream>>>(Abuf, Bbuf, Dbuf, alpha, out);
}
